// KernelNNBoundary_91164975825182
// MI455X (gfx1250) — matrix operand order repaired and checked by reading, not compiled, not run on hardware
//
#include <hip/hip_runtime.h>

typedef __attribute__((ext_vector_type(16))) __bf16 v16bf;
typedef __attribute__((ext_vector_type(8)))  float  v8f;

#define NND   10000
#define NE    20000
#define NEB   4000
#define WID   64
#define KW    128
#define DEPTH 4

#ifndef USE_ASYNC_LDS
#define USE_ASYNC_LDS 1
#endif

// ---------------------------------------------------------------------------
// fc1: x0[n,o] = x[n]*fc1_w[o] + fc1_b[o]; also seeds h = x0
// ---------------------------------------------------------------------------
__global__ void fc1_kernel(const float* __restrict__ x, const float* __restrict__ w,
                           const float* __restrict__ b, float* __restrict__ x0,
                           float* __restrict__ h) {
    int g = blockIdx.x * blockDim.x + threadIdx.x;
    if (g >= NND * WID) return;
    int n = g >> 6, o = g & 63;
    float v = x[n] * w[o] + b[o];
    x0[g] = v;
    h[g] = v;
}

__global__ void zero_f_kernel(float* p, int n) {
    int g = blockIdx.x * blockDim.x + threadIdx.x;
    if (g < n) p[g] = 0.0f;
}
__global__ void zero_i_kernel(int* p, int n) {
    int g = blockIdx.x * blockDim.x + threadIdx.x;
    if (g < n) p[g] = 0;
}

// ---------------------------------------------------------------------------
// degree count + inverse (layer-invariant)
// ---------------------------------------------------------------------------
__global__ void count_kernel(const int* __restrict__ eidx, int nedge, int* __restrict__ deg) {
    int e = blockIdx.x * blockDim.x + threadIdx.x;
    if (e < nedge) atomicAdd(&deg[eidx[nedge + e]], 1);
}
__global__ void deginv_kernel(const int* __restrict__ deg, float* __restrict__ dinv) {
    int g = blockIdx.x * blockDim.x + threadIdx.x;
    if (g < NND) {
        int d = deg[g];
        dinv[g] = 1.0f / (float)(d > 0 ? d : 1);
    }
}

// ---------------------------------------------------------------------------
// edge-attr MLP front (once per edge set): h2 = relu(relu(ea·w1T+b1)·w2T+b2)
// ---------------------------------------------------------------------------
__global__ void edge_mlp_kernel(const float* __restrict__ ea,
                                const float* __restrict__ w1, const float* __restrict__ b1,
                                const float* __restrict__ w2, const float* __restrict__ b2,
                                float* __restrict__ h2out) {
    __shared__ float h1[64];
    __shared__ float eav[6];
    int e = blockIdx.x, t = threadIdx.x;
    if (t < 6) eav[t] = ea[e * 6 + t];
    __syncthreads();
    if (t < 64) {
        float a = b1[t];
        #pragma unroll
        for (int j = 0; j < 6; ++j) a += eav[j] * w1[t * 6 + j];
        h1[t] = a > 0.f ? a : 0.f;
    }
    __syncthreads();
    float a = b2[t];
    for (int j = 0; j < 64; ++j) a += h1[j] * w2[t * 64 + j];
    h2out[e * KW + t] = a > 0.f ? a : 0.f;
}

// ---------------------------------------------------------------------------
// pack h2 tiles into WMMA 16-bit A layout (bf16)
// A layout: M = lane&15;  K = k0*32 + (lane<16?0:8) + (h<8 ? h : h+8)
// ---------------------------------------------------------------------------
__global__ void pack_a_kernel(const float* __restrict__ h2f, int ntiles, __bf16* __restrict__ pack) {
    int g = blockIdx.x * blockDim.x + threadIdx.x;
    if (g >= ntiles * 32) return;
    int tile = g >> 5, lane = g & 31;
    int M = lane & 15;
    int kb = (lane < 16) ? 0 : 8;
    #pragma unroll
    for (int k0 = 0; k0 < 4; ++k0)
        #pragma unroll
        for (int h = 0; h < 16; ++h) {
            int K = k0 * 32 + kb + (h < 8 ? h : h + 8);
            pack[(size_t)(((tile * 4 + k0) * 32 + lane)) * 16 + h] =
                (__bf16)h2f[(size_t)(tile * 16 + M) * KW + K];
        }
}

// ---------------------------------------------------------------------------
// pack w3 into WMMA 16-bit B layout (bf16): B_i[k,n] = w3[(i*64+n)*128 + k]
// index(i,k0,nt,lane) = (i*16 + k0*4 + nt)*32 + lane ; elem h -> K = k0*32 + (lane<16?0:8) + (h<8 ? h : h+8)
// ---------------------------------------------------------------------------
__global__ void pack_b_kernel(const float* __restrict__ w3, __bf16* __restrict__ pack) {
    int g = blockIdx.x * blockDim.x + threadIdx.x;
    if (g >= 64 * 4 * 4 * 32) return;
    int lane = g & 31;
    int nt   = (g >> 5) & 3;
    int k0   = (g >> 7) & 3;
    int i    = g >> 9;
    int n    = nt * 16 + (lane & 15);
    int kb   = (lane < 16) ? 0 : 8;
    #pragma unroll
    for (int h = 0; h < 16; ++h) {
        int K = k0 * 32 + kb + (h < 8 ? h : h + 8);
        pack[(size_t)g * 16 + h] = (__bf16)w3[(size_t)(i * 64 + n) * KW + K];
    }
}

// ---------------------------------------------------------------------------
// async global -> LDS helpers (CDNA5 ASYNCcnt path)
// ---------------------------------------------------------------------------
__device__ __forceinline__ void async_copy16(unsigned ldsOff, const void* g) {
#if USE_ASYNC_LDS
    asm volatile("global_load_async_to_lds_b128 %0, %1, off"
                 :: "v"(ldsOff), "v"(g) : "memory");
#else
    (void)ldsOff; (void)g;
#endif
}
__device__ __forceinline__ void wait_async0() {
#if USE_ASYNC_LDS
#if defined(__has_builtin) && __has_builtin(__builtin_amdgcn_s_wait_asynccnt)
    __builtin_amdgcn_s_wait_asynccnt(0);
#else
    asm volatile("s_wait_asynccnt 0x0" ::: "memory");
#endif
#endif
}

__device__ __forceinline__ void stage_slab(const __bf16* __restrict__ bpack, int i,
                                           char* sB, int parity, int tid) {
    const uint4* g = (const uint4*)(bpack + (size_t)i * 8192) + tid * 4;
    char* dst = sB + parity * 16384 + tid * 64;
#if USE_ASYNC_LDS
    unsigned sOff = (unsigned)(unsigned long long)(void*)dst;  // flat low 32 = LDS offset
    async_copy16(sOff,      g);
    async_copy16(sOff + 16, g + 1);
    async_copy16(sOff + 32, g + 2);
    async_copy16(sOff + 48, g + 3);
#else
    uint4* s = (uint4*)dst;
    s[0] = g[0]; s[1] = g[1]; s[2] = g[2]; s[3] = g[3];
#endif
}

// ---------------------------------------------------------------------------
// hot kernel: per 16-edge tile (one wave32 each, 8 waves/WG):
//   msg[e,o] = sum_i x[src[e],i] * ( (H2 @ B_i)[e,o] + b3[i*64+o] )
// B_i slab (16KB) is double-buffer staged in LDS (async DMA) and shared by
// all 8 waves -> 8x less L2 traffic. 1024 WMMA / tile.
// dyn LDS: [0,32K) B slabs | [32K,64K) xT | [64K,80K) b3 | [80K, +512) dinv
// ---------------------------------------------------------------------------
__global__ void __launch_bounds__(256)
edge_wmma_kernel(const float* __restrict__ hcur,
                 const __bf16* __restrict__ h2pack,
                 const __bf16* __restrict__ bpack,
                 const int* __restrict__ eidx, int nedge,
                 const float* __restrict__ b3,
                 const float* __restrict__ dinv,
                 float* __restrict__ agg) {
    extern __shared__ char smem[];
    char*  sB  = smem;                          // 2 x 16KB
    float* sxT = (float*)(smem + 32768);        // 8 x 1024 f32
    float* sb3 = (float*)(smem + 65536);        // 4096 f32
    float* sd  = (float*)(smem + 81920);        // 8 x 16 f32

    const int tid  = threadIdx.x;
    const int lane = tid & 31;
    const int wv   = tid >> 5;
    const int ntiles = nedge >> 4;
    const int tile = blockIdx.x * 8 + wv;
    const bool active = (tile < ntiles);

    for (int i = tid; i < 4096; i += 256) sb3[i] = b3[i];

    int e0 = tile << 4;
    v16bf a0, a1, a2, a3;
    if (active) {
        for (int idx = lane; idx < 16 * 64; idx += 32) {
            int m = idx >> 6, f = idx & 63;
            int src = eidx[e0 + m];
            sxT[wv * 1024 + f * 16 + m] = hcur[src * WID + f];
        }
        if (lane < 16) sd[wv * 16 + lane] = dinv[eidx[nedge + e0 + lane]];
        const v16bf* h2p = (const v16bf*)h2pack;
        a0 = h2p[(size_t)(tile * 4 + 0) * 32 + lane];
        a1 = h2p[(size_t)(tile * 4 + 1) * 32 + lane];
        a2 = h2p[(size_t)(tile * 4 + 2) * 32 + lane];
        a3 = h2p[(size_t)(tile * 4 + 3) * 32 + lane];
    }

    // prime the pipeline: slab for i=0 into buffer 0
    stage_slab(bpack, 0, sB, 0, tid);

    float acc[4][8];
    #pragma unroll
    for (int nt = 0; nt < 4; ++nt)
        #pragma unroll
        for (int r = 0; r < 8; ++r) acc[nt][r] = 0.f;

    const int mo = (lane < 16) ? 0 : 8;
    const int nc = lane & 15;

    for (int i = 0; i < 64; ++i) {
        wait_async0();
        __syncthreads();                       // slab i resident & prev reads done
        if (i < 63) stage_slab(bpack, i + 1, sB, (i + 1) & 1, tid);
        if (active) {
            const v16bf* sbp = (const v16bf*)(sB + (i & 1) * 16384);
            float xs[8];
            #pragma unroll
            for (int r = 0; r < 8; ++r) xs[r] = sxT[wv * 1024 + i * 16 + mo + r];
            #pragma unroll
            for (int nt = 0; nt < 4; ++nt) {
                v8f c = {};
                c = __builtin_amdgcn_wmma_f32_16x16x32_bf16(false, a0, false, sbp[(0 * 4 + nt) * 32 + lane], (short)0, c, false, false);
                c = __builtin_amdgcn_wmma_f32_16x16x32_bf16(false, a1, false, sbp[(1 * 4 + nt) * 32 + lane], (short)0, c, false, false);
                c = __builtin_amdgcn_wmma_f32_16x16x32_bf16(false, a2, false, sbp[(2 * 4 + nt) * 32 + lane], (short)0, c, false, false);
                c = __builtin_amdgcn_wmma_f32_16x16x32_bf16(false, a3, false, sbp[(3 * 4 + nt) * 32 + lane], (short)0, c, false, false);
                float bv = sb3[i * 64 + nt * 16 + nc];
                #pragma unroll
                for (int r = 0; r < 8; ++r) acc[nt][r] += xs[r] * (c[r] + bv);
            }
        }
    }

    if (!active) return;

    // scatter (pre-scaled mean) with native f32 atomics
    #pragma unroll
    for (int r = 0; r < 8; ++r) {
        int m = mo + r;
        int dst = eidx[nedge + e0 + m];
        float ds = sd[wv * 16 + m];
        #pragma unroll
        for (int nt = 0; nt < 4; ++nt) {
            float v = acc[nt][r] * ds;
            __hip_atomic_fetch_add(&agg[(size_t)dst * WID + nt * 16 + nc], v,
                                   __ATOMIC_RELAXED, __HIP_MEMORY_SCOPE_AGENT);
        }
    }
}

// ---------------------------------------------------------------------------
// node update: hnext = relu(agg + h@(root1+root2) + bias1 + bias2) + x0
// ---------------------------------------------------------------------------
__global__ void node_update_kernel(const float* __restrict__ agg, const float* __restrict__ hcur,
                                   const float* __restrict__ root1, const float* __restrict__ bias1,
                                   const float* __restrict__ root2, const float* __restrict__ bias2,
                                   const float* __restrict__ x0, float* __restrict__ hnext) {
    __shared__ float hrow[64];
    int n = blockIdx.x, o = threadIdx.x;
    hrow[o] = hcur[n * WID + o];
    __syncthreads();
    float a = agg[n * WID + o] + bias1[o] + bias2[o];
    for (int j = 0; j < 64; ++j) a += hrow[j] * (root1[j * 64 + o] + root2[j * 64 + o]);
    float r = a > 0.f ? a : 0.f;
    hnext[n * WID + o] = r + x0[n * WID + o];
}

// ---------------------------------------------------------------------------
// head: out[n] = fc3( relu(fc2(h)) )
// ---------------------------------------------------------------------------
__global__ void head_kernel(const float* __restrict__ h,
                            const float* __restrict__ fc2w, const float* __restrict__ fc2b,
                            const float* __restrict__ fc3w, const float* __restrict__ fc3b,
                            float* __restrict__ out) {
    __shared__ float hrow[64];
    __shared__ float part[128];
    int n = blockIdx.x, t = threadIdx.x;
    if (t < 64) hrow[t] = h[n * WID + t];
    __syncthreads();
    float a = fc2b[t];
    for (int j = 0; j < 64; ++j) a += hrow[j] * fc2w[t * 64 + j];
    a = a > 0.f ? a : 0.f;
    part[t] = a * fc3w[t];
    __syncthreads();
    for (int s = 64; s > 0; s >>= 1) {
        if (t < s) part[t] += part[t + s];
        __syncthreads();
    }
    if (t == 0) out[n] = part[0] + fc3b[0];
}

// ---------------------------------------------------------------------------
extern "C" void kernel_launch(void* const* d_in, const int* in_sizes, int n_in,
                              void* d_out, int out_size, void* d_ws, size_t ws_size,
                              hipStream_t stream) {
    (void)in_sizes; (void)n_in; (void)out_size; (void)ws_size;
    const float* x      = (const float*)d_in[0];
    const int*   eiA    = (const int*)d_in[1];
    const float* eaA    = (const float*)d_in[2];
    const int*   eiB    = (const int*)d_in[3];
    const float* eaB    = (const float*)d_in[4];
    const float* fc1w   = (const float*)d_in[5];
    const float* fc1b   = (const float*)d_in[6];
    const float* fc2w   = (const float*)d_in[7];
    const float* fc2b   = (const float*)d_in[8];
    const float* fc3w   = (const float*)d_in[9];
    const float* fc3b   = (const float*)d_in[10];
    const float* k1w1   = (const float*)d_in[11];
    const float* k1b1   = (const float*)d_in[12];
    const float* k1w2   = (const float*)d_in[13];
    const float* k1b2   = (const float*)d_in[14];
    const float* k1w3   = (const float*)d_in[15];
    const float* k1b3   = (const float*)d_in[16];
    const float* root1  = (const float*)d_in[17];
    const float* bias1  = (const float*)d_in[18];
    const float* k2w1   = (const float*)d_in[19];
    const float* k2b1   = (const float*)d_in[20];
    const float* k2w2   = (const float*)d_in[21];
    const float* k2b2   = (const float*)d_in[22];
    const float* k2w3   = (const float*)d_in[23];
    const float* k2b3   = (const float*)d_in[24];
    const float* root2  = (const float*)d_in[25];
    const float* bias2  = (const float*)d_in[26];
    float* out = (float*)d_out;

    char* ws = (char*)d_ws;
    size_t off = 0;
    auto alloc = [&](size_t bytes) -> char* {
        char* p = ws + off;
        off += (bytes + 255) & ~(size_t)255;
        return p;
    };
    float*  x0    = (float*)alloc((size_t)NND * WID * 4);
    float*  hA    = (float*)alloc((size_t)NND * WID * 4);
    float*  hB    = (float*)alloc((size_t)NND * WID * 4);
    float*  agg   = (float*)alloc((size_t)NND * WID * 4);
    float*  h2fA  = (float*)alloc((size_t)NE * KW * 4);
    float*  h2fB  = (float*)alloc((size_t)NEB * KW * 4);
    __bf16* packA = (__bf16*)alloc((size_t)(NE / 16) * 4 * 32 * 16 * 2);
    __bf16* packB = (__bf16*)alloc((size_t)(NEB / 16) * 4 * 32 * 16 * 2);
    __bf16* bpk1  = (__bf16*)alloc((size_t)64 * 4 * 4 * 32 * 16 * 2);
    __bf16* bpk2  = (__bf16*)alloc((size_t)64 * 4 * 4 * 32 * 16 * 2);
    int*    degA  = (int*)alloc((size_t)NND * 4);
    int*    degB  = (int*)alloc((size_t)NND * 4);
    float*  dinvA = (float*)alloc((size_t)NND * 4);
    float*  dinvB = (float*)alloc((size_t)NND * 4);

    const int TA = NE / 16, TB = NEB / 16;
    const size_t EDGE_LDS = 32768 + 32768 + 16384 + 512;

    // ---- once-per-call precompute ----
    fc1_kernel<<<(NND * WID + 255) / 256, 256, 0, stream>>>(x, fc1w, fc1b, x0, hA);

    zero_i_kernel<<<(NND + 255) / 256, 256, 0, stream>>>(degA, NND);
    zero_i_kernel<<<(NND + 255) / 256, 256, 0, stream>>>(degB, NND);
    count_kernel<<<(NE + 255) / 256, 256, 0, stream>>>(eiA, NE, degA);
    count_kernel<<<(NEB + 255) / 256, 256, 0, stream>>>(eiB, NEB, degB);
    deginv_kernel<<<(NND + 255) / 256, 256, 0, stream>>>(degA, dinvA);
    deginv_kernel<<<(NND + 255) / 256, 256, 0, stream>>>(degB, dinvB);

    edge_mlp_kernel<<<NE, 128, 0, stream>>>(eaA, k1w1, k1b1, k1w2, k1b2, h2fA);
    edge_mlp_kernel<<<NEB, 128, 0, stream>>>(eaB, k2w1, k2b1, k2w2, k2b2, h2fB);
    pack_a_kernel<<<(TA * 32 + 255) / 256, 256, 0, stream>>>(h2fA, TA, packA);
    pack_a_kernel<<<(TB * 32 + 255) / 256, 256, 0, stream>>>(h2fB, TB, packB);
    pack_b_kernel<<<(64 * 4 * 4 * 32 + 255) / 256, 256, 0, stream>>>(k1w3, bpk1);
    pack_b_kernel<<<(64 * 4 * 4 * 32 + 255) / 256, 256, 0, stream>>>(k2w3, bpk2);

    // ---- depth loop ----
    for (int l = 0; l < DEPTH; ++l) {
        float* hin  = (l & 1) ? hB : hA;
        float* hout = (l & 1) ? hA : hB;
        zero_f_kernel<<<(NND * WID + 255) / 256, 256, 0, stream>>>(agg, NND * WID);
        edge_wmma_kernel<<<(TA + 7) / 8, 256, EDGE_LDS, stream>>>(hin, packA, bpk1, eiA, NE,
                                                                  k1b3, dinvA, agg);
        edge_wmma_kernel<<<(TB + 7) / 8, 256, EDGE_LDS, stream>>>(hin, packB, bpk2, eiB, NEB,
                                                                  k2b3, dinvB, agg);
        node_update_kernel<<<NND, 64, 0, stream>>>(agg, hin, root1, bias1, root2, bias2,
                                                   x0, hout);
    }
    // final h after 4 layers is in hA (A->B->A->B->A)
    head_kernel<<<NND, 128, 0, stream>>>(hA, fc2w, fc2b, fc3w, fc3b, out);
}
